// EMAQuantizer_55722905699288
// MI455X (gfx1250) — matrix operand order repaired and checked by reading, not compiled, not run on hardware
//
#include <hip/hip_runtime.h>

// ---------------------------------------------------------------------------
// VQ codebook assignment (EMAQuantizer forward) for MI455X / gfx1250.
//   z:         [4, 64, 32, 32, 32] f32   (d_in[0])
//   embedding: [1024, 64] f32            (d_in[1])
// Outputs (concatenated f32 in d_out):
//   z_q [4,64,32,32,32] | loss | perplexity | indices [4,32,32,32] | mean(dist)
//
// Distance GEMM runs on the bf16 WMMA pipe using a bf16x3 (hi/lo) split for
// f32-class accuracy: z.e = zhi.ehi + zlo.ehi + zhi.elo  (lo*lo dropped).
// Codebook is pre-converted to a ready-made LDS image and staged per-WG with
// CDNA5 async global->LDS b128 copies (ASYNCcnt).
// ---------------------------------------------------------------------------

typedef __attribute__((ext_vector_type(2)))  float    v2f;
typedef __attribute__((ext_vector_type(4)))  float    v4f;
typedef __attribute__((ext_vector_type(8)))  float    v8f;
typedef __attribute__((ext_vector_type(4)))  unsigned v4u;
typedef __attribute__((ext_vector_type(16))) __bf16   v16bf;

#define N_TOK      131072          // 4*32*32*32 tokens
#define K_CODE     1024
#define CCH        64
#define SPAT       32768           // 32*32*32
#define TOK_PER_WG 512

// LDS image layout (in dwords): bf16 codebook hi-plane, lo-plane, f32 0.5*||e||^2
#define ROW_DW   36                // 72 halves per row (64 data + 8 pad) -> conflict-free b128
#define LDS_HI   0
#define LDS_LO   (K_CODE * ROW_DW)             // 36864
#define LDS_EN   (2 * K_CODE * ROW_DW)         // 73728
#define LDS_DW   (LDS_EN + K_CODE)             // 74752 dwords = 299008 B (< 320 KB WGP LDS)

// d_out float offsets
#define OUT_ZQ   0
#define OUT_LOSS 8388608
#define OUT_PERP 8388609
#define OUT_IDX  8388610
#define OUT_MD   8519682

static __device__ __forceinline__ v8f wmma_bf16(v16bf a, v16bf b, v8f c) {
  return __builtin_amdgcn_wmma_f32_16x16x32_bf16(false, a, false, b, (short)0, c,
                                                 false, false);
}

static __device__ __forceinline__ unsigned pack2bf(float a, float b) {
  union { __bf16 h[2]; unsigned u; } p;
  p.h[0] = (__bf16)a;               // even K -> low half
  p.h[1] = (__bf16)b;               // odd  K -> high half
  return p.u;
}

// ---------------------------------------------------------------------------
// Workspace zero-init (harness does not re-zero between replays)
// ---------------------------------------------------------------------------
__global__ void zero_ws(unsigned* counts, float* sum_z, float* sum_e, float* scal) {
  const int t = threadIdx.x;            // 1024 threads, 1 block
  counts[t] = 0u;
  if (t < 64) { sum_z[t] = 0.f; sum_e[t] = 0.f; }
  if (t < 2)  scal[t] = 0.f;
}

// ---------------------------------------------------------------------------
// Codebook prep: build the packed-bf16 hi/lo LDS image (row stride 36 dwords)
// plus f32 0.5*||e_k||^2, and accumulate total ||e||^2.
// ---------------------------------------------------------------------------
__global__ void prep_codebook(const float* __restrict__ emb, unsigned* __restrict__ blob,
                              float* __restrict__ scal) {
  const int k = blockIdx.x * 256 + threadIdx.x;   // grid = 4 blocks x 256
  const v2f* e2 = (const v2f*)emb + k * 32;
  float nrm = 0.f;
  #pragma unroll 8
  for (int d = 0; d < 32; ++d) {                  // dword-pair within row
    const v2f ev = e2[d];
    const __bf16 h0 = (__bf16)ev.x, h1 = (__bf16)ev.y;
    const float  l0 = ev.x - (float)h0, l1 = ev.y - (float)h1;
    blob[LDS_HI + k * ROW_DW + d] = pack2bf((float)h0, (float)h1);
    blob[LDS_LO + k * ROW_DW + d] = pack2bf(l0, l1);
    nrm += ev.x * ev.x + ev.y * ev.y;
  }
  blob[LDS_EN + k] = __builtin_bit_cast(unsigned, 0.5f * nrm);
  atomicAdd(&scal[1], nrm);                       // sum_e2
}

__global__ void col_sums(const float* __restrict__ emb, float* __restrict__ sum_e) {
  const int ch = threadIdx.x;                     // 64 threads, 1 block
  float s = 0.f;
  for (int k = 0; k < K_CODE; ++k) s += emb[k * CCH + ch];
  sum_e[ch] = s;
}

// ---------------------------------------------------------------------------
// z statistics for closed-form mean(dist): per-channel sums of z, sum of z^2.
// Each thread handles 4 contiguous elements (b128); a block covers 1024
// aligned elements -> single channel per block.
// ---------------------------------------------------------------------------
__global__ void __launch_bounds__(256) z_stats(const float* __restrict__ z,
                                               float* __restrict__ sum_z,
                                               float* __restrict__ scal) {
  __shared__ float r1[8], r2[8];
  const int i = (blockIdx.x * 256 + threadIdx.x) * 4;
  const v4f v = *(const v4f*)(z + i);
  float s1 = v.x + v.y + v.z + v.w;
  float s2 = v.x * v.x + v.y * v.y + v.z * v.z + v.w * v.w;
  #pragma unroll
  for (int off = 16; off >= 1; off >>= 1) {
    s1 += __shfl_down(s1, off, 32);
    s2 += __shfl_down(s2, off, 32);
  }
  const int wave = threadIdx.x >> 5, lane = threadIdx.x & 31;
  if (lane == 0) { r1[wave] = s1; r2[wave] = s2; }
  __syncthreads();
  if (threadIdx.x == 0) {
    float t1 = 0.f, t2 = 0.f;
    #pragma unroll
    for (int w = 0; w < 8; ++w) { t1 += r1[w]; t2 += r2[w]; }
    const int ch = (i >> 15) & 63;
    atomicAdd(&sum_z[ch], t1);
    atomicAdd(&scal[0], t2);                      // sum_z2
  }
}

// ---------------------------------------------------------------------------
// Main kernel: bf16x3 WMMA distance GEMM + fused argmin.
//   - codebook image async-DMA'd into LDS (global_load_async_to_lds_b128)
//   - each wave: 2 register-blocked 16-token tiles; 64 code-tiles of 16
//   - per code-tile: 8 b128 DS loads feed 12 v_wmma_f32_16x16x32_bf16
//   argmin(dist) == argmax(z.e - 0.5||e||^2)
// ---------------------------------------------------------------------------
__global__ void __launch_bounds__(256) vq_argmin_wmma(
    const float* __restrict__ z, const unsigned* __restrict__ blob,
    int* __restrict__ ws_idx, float* __restrict__ out_idx_f) {
  extern __shared__ unsigned ldsu[];
  const int tid = threadIdx.x;

  // ---- Async-stage the prebuilt codebook image into LDS (ASYNCcnt) ----
  {
    const unsigned long long src = (unsigned long long)blob;
    for (int i = tid; i < LDS_DW / 4; i += 256) {
      const unsigned ldsoff = (unsigned)i * 16u;              // dynamic LDS starts at 0
      const unsigned long long ga = src + (unsigned long long)i * 16ull;
      asm volatile("global_load_async_to_lds_b128 %0, %1, off"
                   :: "v"(ldsoff), "v"(ga) : "memory");
    }
    asm volatile("s_wait_asynccnt 0x0" ::: "memory");
  }
  __syncthreads();

  const int wave = tid >> 5, lane = tid & 31;
  const int m = lane & 15;           // M / N index within tile
  const int g = lane >> 4;           // half-wave group

  for (int pass = 0; pass < 2; ++pass) {
    const int tile0 = pass * 16 + wave * 2;     // two adjacent tiles per wave
    const int tok_base = blockIdx.x * TOK_PER_WG + tile0 * 16;

    // ---- Load A (2 tiles x 16 tokens x 64 ch), split hi/lo, WMMA layout ----
    // A k order used here (contiguous, same as B): VGPR v -> K=2v+16g(,+1)
    v16bf ahi[2][2], alo[2][2];                 // [tile][chunk]
    #pragma unroll
    for (int u = 0; u < 2; ++u) {
      const int tok = tok_base + u * 16 + m;
      const int bb  = tok >> 15;
      const int s   = tok & (SPAT - 1);
      const float* zp = z + ((size_t)bb * CCH) * SPAT + s;
      #pragma unroll
      for (int c = 0; c < 2; ++c) {
        #pragma unroll
        for (int v = 0; v < 8; ++v) {
          const int kloc = 2 * v + 16 * g;
          #pragma unroll
          for (int p = 0; p < 2; ++p) {
            const int ch = 32 * c + kloc + p;
            const float zv = zp[(size_t)ch * SPAT];
            const __bf16 h = (__bf16)zv;
            ahi[u][c][2 * v + p] = h;
            alo[u][c][2 * v + p] = (__bf16)(zv - (float)h);
          }
        }
      }
    }

    float bestv[2][8];
    int   bestn[2][8];
    #pragma unroll
    for (int u = 0; u < 2; ++u)
      #pragma unroll
      for (int j = 0; j < 8; ++j) { bestv[u][j] = -3.4e38f; bestn[u][j] = 0; }

    for (int n0 = 0; n0 < K_CODE; n0 += 16) {
      const int n = n0 + m;                     // this lane's code row for B
      v8f acc[2] = {{}, {}};
      #pragma unroll
      for (int c = 0; c < 2; ++c) {
        // B 32x16: lanes0-15 K=0-15, lanes16-31 K=16-31; contiguous 16 halves
        const int base = n * ROW_DW + 16 * c + 8 * g;     // dword offset (16B aligned)
        union { v4u w[2]; v16bf b; } Uh, Ul;
        const v4u* ph = (const v4u*)(ldsu + LDS_HI + base);
        const v4u* pl = (const v4u*)(ldsu + LDS_LO + base);
        Uh.w[0] = ph[0]; Uh.w[1] = ph[1];
        Ul.w[0] = pl[0]; Ul.w[1] = pl[1];
        const v16bf Bhi = Uh.b, Blo = Ul.b;
        #pragma unroll
        for (int u = 0; u < 2; ++u) {
          acc[u] = wmma_bf16(ahi[u][c], Bhi, acc[u]);
          acc[u] = wmma_bf16(alo[u][c], Bhi, acc[u]);
          acc[u] = wmma_bf16(ahi[u][c], Blo, acc[u]);
        }
      }
      const float eh = __builtin_bit_cast(float, ldsu[LDS_EN + n]); // 0.5*||e_n||^2
      #pragma unroll
      for (int u = 0; u < 2; ++u)
        #pragma unroll
        for (int j = 0; j < 8; ++j) {
          const float sc = acc[u][j] - eh;
          if (sc > bestv[u][j]) { bestv[u][j] = sc; bestn[u][j] = n; }
        }
    }

    // Reduce argmax across the 16 lanes of each half-wave (ties -> lowest index,
    // matching jnp.argmin). xor offsets 1..8 stay within each 16-lane group.
    #pragma unroll
    for (int off = 8; off >= 1; off >>= 1) {
      #pragma unroll
      for (int u = 0; u < 2; ++u)
        #pragma unroll
        for (int j = 0; j < 8; ++j) {
          const float ov = __shfl_xor(bestv[u][j], off, 32);
          const int   on = __shfl_xor(bestn[u][j], off, 32);
          if (ov > bestv[u][j] || (ov == bestv[u][j] && on < bestn[u][j])) {
            bestv[u][j] = ov; bestn[u][j] = on;
          }
        }
    }
    if (m == 0) {
      #pragma unroll
      for (int u = 0; u < 2; ++u)
        #pragma unroll
        for (int j = 0; j < 8; ++j) {           // slot j = token j + 8*g
          const int t = tok_base + u * 16 + j + 8 * g;
          ws_idx[t]    = bestn[u][j];
          out_idx_f[t] = (float)bestn[u][j];
        }
    }
  }
}

// ---------------------------------------------------------------------------
// z_q gather: 4 tokens per thread, b128 stores; codebook gather (256 KB)
// stays resident in cache.
// ---------------------------------------------------------------------------
__global__ void __launch_bounds__(256) zq_gather(const float* __restrict__ emb,
                                                 const int* __restrict__ ws_idx,
                                                 float* __restrict__ out_zq) {
  const int i   = (blockIdx.x * 256 + threadIdx.x) * 4;
  const int bb  = i >> 21;                        // / (64*32768)
  const int rem = i & ((1 << 21) - 1);
  const int ch  = rem >> 15;
  const int s   = rem & (SPAT - 1);
  const int n   = (bb << 15) | s;
  v4f r;
  r.x = emb[ws_idx[n + 0] * CCH + ch];
  r.y = emb[ws_idx[n + 1] * CCH + ch];
  r.z = emb[ws_idx[n + 2] * CCH + ch];
  r.w = emb[ws_idx[n + 3] * CCH + ch];
  *(v4f*)(out_zq + i) = r;
}

__global__ void histo(const int* __restrict__ ws_idx, unsigned* __restrict__ counts) {
  const int n = blockIdx.x * 256 + threadIdx.x;
  atomicAdd(&counts[ws_idx[n]], 1u);
}

// ---------------------------------------------------------------------------
// Perplexity + mean(dist) + loss
// mean(dist) = mean||z||^2 + mean||e||^2 - (2/NK) (sum z).(sum e)
// ---------------------------------------------------------------------------
__global__ void finalize(const unsigned* __restrict__ counts, const float* __restrict__ sum_z,
                         const float* __restrict__ sum_e, const float* __restrict__ scal,
                         float* __restrict__ out) {
  __shared__ float red[256];
  const int t = threadIdx.x;
  float h = 0.f;
  for (int k = t; k < K_CODE; k += 256) {
    const float p = (float)counts[k] / (float)N_TOK;
    h -= p * logf(p + 1e-10f);
  }
  red[t] = h;
  __syncthreads();
  for (int off = 128; off >= 1; off >>= 1) {
    if (t < off) red[t] += red[t + off];
    __syncthreads();
  }
  if (t == 0) {
    float dot = 0.f;
    #pragma unroll 8
    for (int ch = 0; ch < CCH; ++ch) dot += sum_z[ch] * sum_e[ch];
    const float md = scal[0] / (float)N_TOK + scal[1] / (float)K_CODE
                   - 2.f * dot / ((float)N_TOK * (float)K_CODE);
    out[OUT_LOSS] = 0.f;
    out[OUT_PERP] = expf(red[0]);
    out[OUT_MD]   = md;
  }
}

// ---------------------------------------------------------------------------
extern "C" void kernel_launch(void* const* d_in, const int* in_sizes, int n_in,
                              void* d_out, int out_size, void* d_ws, size_t ws_size,
                              hipStream_t stream) {
  const float* z   = (const float*)d_in[0];
  const float* emb = (const float*)d_in[1];
  float* out = (float*)d_out;

  // Workspace layout
  int*      ws_idx = (int*)d_ws;                             // 131072 ints (512 KB)
  unsigned* blob   = (unsigned*)((char*)d_ws + 524288);      // LDS image, 299008 B
  unsigned* counts = (unsigned*)((char*)d_ws + 823296);      // 1024 uints
  float*    sum_z  = (float*)((char*)d_ws + 827392);         // 64 floats
  float*    sum_e  = (float*)((char*)d_ws + 827648);         // 64 floats
  float*    scal   = (float*)((char*)d_ws + 827904);         // [0]=sum_z2 [1]=sum_e2

  zero_ws<<<1, 1024, 0, stream>>>(counts, sum_z, sum_e, scal);
  prep_codebook<<<K_CODE / 256, 256, 0, stream>>>(emb, blob, scal);
  col_sums<<<1, 64, 0, stream>>>(emb, sum_e);
  z_stats<<<(N_TOK * CCH) / 1024, 256, 0, stream>>>(z, sum_z, scal);

  vq_argmin_wmma<<<N_TOK / TOK_PER_WG, 256, LDS_DW * sizeof(unsigned), stream>>>(
      z, blob, ws_idx, out + OUT_IDX);

  zq_gather<<<(N_TOK * CCH) / 1024, 256, 0, stream>>>(emb, ws_idx, out + OUT_ZQ);
  histo<<<N_TOK / 256, 256, 0, stream>>>(ws_idx, counts);
  finalize<<<1, 256, 0, stream>>>(counts, sum_z, sum_e, scal, out);
}
